// BNLSTM_24275155157039
// MI455X (gfx1250) — matrix operand order repaired and checked by reading, not compiled, not run on hardware
//
#include <hip/hip_runtime.h>

// ---------------------------------------------------------------------------
// BN-LSTM for MI455X (gfx1250, wave32, WMMA).
// GEMMs run on v_wmma_f32_16x16x32_bf16 with a 2-term bf16 split (hi+lo) of
// both operands -> ~fp32 accuracy on the bf16 matrix pipe (3 WMMAs per K=32).
// Splits are precomputed: weights are packed ONCE into WMMA-fragment-order
// bf16 hi/lo planes; the recurrent h is packed by the h-update kernel itself.
// Only the x-path A fragment is split inline. All batch-stat reductions use
// fixed-slot partials (no float atomics) -> bitwise-deterministic replays.
// ---------------------------------------------------------------------------

typedef __attribute__((ext_vector_type(16))) __bf16        v16bf;
typedef __attribute__((ext_vector_type(8)))  float         v8f;
typedef __attribute__((ext_vector_type(8)))  unsigned int  v8u;

#define BB 512   // batch
#define TT 128   // timesteps
#define FF 512   // input features
#define HH 512   // hidden
#define G4 2048  // 4*H
#define KD 512   // K dim of both GEMMs
#define KP 256   // KD/2 packed k-pairs
#define EPS 1e-5f

__device__ __forceinline__ unsigned int bf_hi_bits(float x) {
    return __float_as_uint(x) >> 16;                 // truncation -> exact residual
}
__device__ __forceinline__ unsigned int bf_lo_bits(float x) {
    float hi = __uint_as_float(__float_as_uint(x) & 0xFFFF0000u);
    unsigned u = __float_as_uint(x - hi);
    u += 0x7FFFu + ((u >> 16) & 1u);                 // RNE
    return u >> 16;
}
__device__ __forceinline__ v16bf mkfrag(uint4 a, uint4 b) {
    v8u t;
    t[0] = a.x; t[1] = a.y; t[2] = a.z; t[3] = a.w;
    t[4] = b.x; t[5] = b.y; t[6] = b.z; t[7] = b.w;
    return __builtin_bit_cast(v16bf, t);
}

// ---------------------------------------------------------------------------
// One-time weight pack: W[KD][G4] f32 -> hi/lo planes in B-fragment order:
//   p[col*KP + k/2] = (bf16(W[k+1][col]) << 16) | bf16(W[k][col])
// ---------------------------------------------------------------------------
__global__ __launch_bounds__(256) void bnlstm_packw(
    const float* __restrict__ W,
    unsigned int* __restrict__ p_hi, unsigned int* __restrict__ p_lo)
{
    int tid = blockIdx.x * blockDim.x + threadIdx.x;   // KP*G4 threads
    if (tid >= KP * G4) return;
    int col = tid & (G4 - 1);
    int kp  = tid >> 11;
    float w0 = W[(size_t)(2 * kp) * G4 + col];
    float w1 = W[(size_t)(2 * kp + 1) * G4 + col];
    p_hi[(size_t)col * KP + kp] = (bf_hi_bits(w1) << 16) | bf_hi_bits(w0);
    p_lo[(size_t)col * KP + kp] = (bf_lo_bits(w1) << 16) | bf_lo_bits(w0);
}

// ---------------------------------------------------------------------------
// Dual GEMM: z==0 -> Gh = h @ W_h (A pre-packed bf16 hi/lo),
//            z==1 -> Gi = x_t @ W_i (A split inline from f32).
// Block = 256 thr = 8 waves as 2(M) x 4(N); wave tile = 16 rows x 64 cols.
// B fragments come pre-swizzled -> two uint4 loads per term per tile.
// ---------------------------------------------------------------------------
__global__ __launch_bounds__(256) void bnlstm_gemm(
    const unsigned int* __restrict__ hp_hi, const unsigned int* __restrict__ hp_lo,
    const float* __restrict__ X, int ldax,
    const unsigned int* __restrict__ wph_hi, const unsigned int* __restrict__ wph_lo,
    const unsigned int* __restrict__ wpi_hi, const unsigned int* __restrict__ wpi_lo,
    float* __restrict__ Ch, float* __restrict__ Ci,
    float* __restrict__ psum_h, float* __restrict__ psq_h,
    float* __restrict__ psum_i, float* __restrict__ psq_i)
{
    const bool xpath = (blockIdx.z != 0);
    const unsigned int* wb_hi = xpath ? wpi_hi : wph_hi;
    const unsigned int* wb_lo = xpath ? wpi_lo : wph_lo;
    float* C  = xpath ? Ci : Ch;
    float* ps = xpath ? psum_i : psum_h;
    float* pq = xpath ? psq_i  : psq_h;

    const int lane = threadIdx.x & 31;
    const int wave = threadIdx.x >> 5;
    const int wm = wave >> 2;          // 0..1
    const int wn = wave & 3;           // 0..3
    const int rowBase = blockIdx.y * 32 + wm * 16;
    const int colBase = blockIdx.x * 256 + wn * 64;
    const int r  = lane & 15;
    const int lh = lane >> 4;

    v8f acc[4];
    acc[0] = v8f{}; acc[1] = v8f{}; acc[2] = v8f{}; acc[3] = v8f{};

    const unsigned int* ah_p = hp_hi + (size_t)(rowBase + r) * KP;
    const unsigned int* al_p = hp_lo + (size_t)(rowBase + r) * KP;
    const float*        xrow = X + (size_t)(rowBase + r) * (size_t)ldax;

    for (int kc = 0; kc < KD; kc += 32) {
        // ---- A fragment: pair i holds K = kc + (i&4?16:0) + lh*8 + (i&3)*2
        v16bf ahi, alo;
        if (!xpath) {
            const int base = (kc >> 1) + lh * 4;     // packed pair index
            uint4 g0 = *(const uint4*)(ah_p + base);
            uint4 g1 = *(const uint4*)(ah_p + base + 8);
            ahi = mkfrag(g0, g1);
            uint4 l0 = *(const uint4*)(al_p + base);
            uint4 l1 = *(const uint4*)(al_p + base + 8);
            alo = mkfrag(l0, l1);
        } else {
            v8u th, tl;
#pragma unroll
            for (int i = 0; i < 8; ++i) {
                int k = kc + ((i & 4) ? 16 : 0) + (lh << 3) + ((i & 3) << 1);
                float2 a2 = *(const float2*)(xrow + k);
                th[i] = (bf_hi_bits(a2.y) << 16) | bf_hi_bits(a2.x);
                tl[i] = (bf_lo_bits(a2.y) << 16) | bf_lo_bits(a2.x);
            }
            ahi = __builtin_bit_cast(v16bf, th);
            alo = __builtin_bit_cast(v16bf, tl);
        }
#pragma unroll
        for (int ti = 0; ti < 4; ++ti) {
            const int col = colBase + ti * 16 + r;
            const size_t base = (size_t)col * KP + (kc >> 1) + lh * 4;
            uint4 h0 = *(const uint4*)(wb_hi + base);
            uint4 h1 = *(const uint4*)(wb_hi + base + 8);
            uint4 l0 = *(const uint4*)(wb_lo + base);
            uint4 l1 = *(const uint4*)(wb_lo + base + 8);
            v16bf bhi = mkfrag(h0, h1);
            v16bf blo = mkfrag(l0, l1);
            acc[ti] = __builtin_amdgcn_wmma_f32_16x16x32_bf16(
                false, ahi, false, bhi, (short)0, acc[ti], false, false);
            acc[ti] = __builtin_amdgcn_wmma_f32_16x16x32_bf16(
                false, ahi, false, blo, (short)0, acc[ti], false, false);
            acc[ti] = __builtin_amdgcn_wmma_f32_16x16x32_bf16(
                false, alo, false, bhi, (short)0, acc[ti], false, false);
        }
    }

    // ---- store C + deterministic per-column partial stats
    const int pslot = blockIdx.y * 2 + wm;           // 0..31 partials per column
#pragma unroll
    for (int ti = 0; ti < 4; ++ti) {
        const int col = colBase + ti * 16 + r;
        float s = 0.f, q = 0.f;
#pragma unroll
        for (int j = 0; j < 8; ++j) {
            float v = acc[ti][j];
            C[(size_t)(rowBase + lh * 8 + j) * G4 + col] = v;
            s += v; q += v * v;
        }
        s += __shfl_xor(s, 16, 32);
        q += __shfl_xor(q, 16, 32);
        if (lh == 0) {
            ps[(size_t)col * 32 + pslot] = s;
            pq[(size_t)col * 32 + pslot] = q;
        }
    }
}

// ---------------------------------------------------------------------------
// Fold 32 partials/column into BN scale/shift for both gate paths.
// ---------------------------------------------------------------------------
__global__ __launch_bounds__(256) void bnlstm_fin_gates(
    const float* __restrict__ psum_h, const float* __restrict__ psq_h,
    const float* __restrict__ psum_i, const float* __restrict__ psq_i,
    const float* __restrict__ gam_h, const float* __restrict__ bet_h,
    const float* __restrict__ gam_i, const float* __restrict__ bet_i,
    float* __restrict__ scale_h, float* __restrict__ shift_h,
    float* __restrict__ scale_i, float* __restrict__ shift_i)
{
    int t = blockIdx.x * blockDim.x + threadIdx.x;   // 0..4095
    const float invB = 1.f / (float)BB;
    if (t < G4) {
        float s = 0.f, q = 0.f;
#pragma unroll 4
        for (int p = 0; p < 32; ++p) { s += psum_h[(size_t)t * 32 + p]; q += psq_h[(size_t)t * 32 + p]; }
        float m  = s * invB;
        float va = q * invB - m * m;
        float sc = gam_h[t] * rsqrtf(va + EPS);
        scale_h[t] = sc;
        shift_h[t] = bet_h[t] - m * sc;
    } else if (t < 2 * G4) {
        t -= G4;
        float s = 0.f, q = 0.f;
#pragma unroll 4
        for (int p = 0; p < 32; ++p) { s += psum_i[(size_t)t * 32 + p]; q += psq_i[(size_t)t * 32 + p]; }
        float m  = s * invB;
        float va = q * invB - m * m;
        float sc = gam_i[t] * rsqrtf(va + EPS);
        scale_i[t] = sc;
        shift_i[t] = bet_i[t] - m * sc;
    }
}

// ---------------------------------------------------------------------------
// Gates -> c1, sigmoid(o); deterministic LDS tree for c1 batch stats.
// block = (64 cols, 4 row-groups), thread does 8 rows. grid = (8, 16).
// ---------------------------------------------------------------------------
__global__ __launch_bounds__(256) void bnlstm_cell(
    const float* __restrict__ Gh, const float* __restrict__ Gi,
    float* __restrict__ c_buf, float* __restrict__ sig_o,
    const float* __restrict__ scale_h, const float* __restrict__ shift_h,
    const float* __restrict__ scale_i, const float* __restrict__ shift_i,
    const float* __restrict__ bias,
    float* __restrict__ psum_c, float* __restrict__ psq_c)
{
    __shared__ float ls[4][64];
    __shared__ float lq[4][64];

    const int tx = threadIdx.x;
    const int ty = threadIdx.y;
    const int j  = blockIdx.x * 64 + tx;
    const int b0 = blockIdx.y * 32 + ty * 8;

    const float shF = scale_h[j],          hhF = shift_h[j];
    const float shI = scale_h[HH + j],     hhI = shift_h[HH + j];
    const float shO = scale_h[2 * HH + j], hhO = shift_h[2 * HH + j];
    const float shG = scale_h[3 * HH + j], hhG = shift_h[3 * HH + j];
    const float siF = scale_i[j],          hiF = shift_i[j];
    const float siI = scale_i[HH + j],     hiI = shift_i[HH + j];
    const float siO = scale_i[2 * HH + j], hiO = shift_i[2 * HH + j];
    const float siG = scale_i[3 * HH + j], hiG = shift_i[3 * HH + j];
    const float bF = bias[j], bI = bias[HH + j], bO = bias[2 * HH + j], bG = bias[3 * HH + j];

    float s = 0.f, q = 0.f;
#pragma unroll
    for (int rr = 0; rr < 8; ++rr) {
        const int b = b0 + rr;
        const size_t gbase = (size_t)b * G4;
        const size_t hbase = (size_t)b * HH + j;
        float f  = Gh[gbase + j]          * shF + hhF + Gi[gbase + j]          * siF + hiF + bF;
        float ii = Gh[gbase + HH + j]     * shI + hhI + Gi[gbase + HH + j]     * siI + hiI + bI;
        float o  = Gh[gbase + 2 * HH + j] * shO + hhO + Gi[gbase + 2 * HH + j] * siO + hiO + bO;
        float g  = Gh[gbase + 3 * HH + j] * shG + hhG + Gi[gbase + 3 * HH + j] * siG + hiG + bG;
        float sf = 1.f / (1.f + expf(-f));
        float si = 1.f / (1.f + expf(-ii));
        float so = 1.f / (1.f + expf(-o));
        float c1 = sf * c_buf[hbase] + si * tanhf(g);
        c_buf[hbase] = c1;
        sig_o[hbase] = so;
        s += c1; q += c1 * c1;
    }
    ls[ty][tx] = s;
    lq[ty][tx] = q;
    __syncthreads();
    if (ty == 0) {
        float S = ls[0][tx] + ls[1][tx] + ls[2][tx] + ls[3][tx];
        float Q = lq[0][tx] + lq[1][tx] + lq[2][tx] + lq[3][tx];
        psum_c[(size_t)j * 16 + blockIdx.y] = S;
        psq_c [(size_t)j * 16 + blockIdx.y] = Q;
    }
}

__global__ __launch_bounds__(256) void bnlstm_fin_c(
    const float* __restrict__ psum_c, const float* __restrict__ psq_c,
    const float* __restrict__ gam_c, const float* __restrict__ bet_c,
    float* __restrict__ scale_c, float* __restrict__ shift_c)
{
    int j = blockIdx.x * blockDim.x + threadIdx.x;
    if (j >= HH) return;
    float s = 0.f, q = 0.f;
#pragma unroll
    for (int p = 0; p < 16; ++p) { s += psum_c[(size_t)j * 16 + p]; q += psq_c[(size_t)j * 16 + p]; }
    const float invB = 1.f / (float)BB;
    float m  = s * invB;
    float va = q * invB - m * m;
    float sc = gam_c[j] * rsqrtf(va + EPS);
    scale_c[j] = sc;
    shift_c[j] = bet_c[j] - m * sc;
}

// ---------------------------------------------------------------------------
// h1 = sig(o)*tanh(bn(c1)); writes f32 h_buf + output slice + packed bf16
// hi/lo A-fragment pairs for the next timestep's h-path GEMM.
// One thread handles a (b, j-pair).
// ---------------------------------------------------------------------------
__global__ __launch_bounds__(256) void bnlstm_hup(
    const float* __restrict__ c_buf, const float* __restrict__ sig_o,
    const float* __restrict__ scale_c, const float* __restrict__ shift_c,
    float* __restrict__ h_buf, unsigned int* __restrict__ hp_hi,
    unsigned int* __restrict__ hp_lo, float* __restrict__ out, int t)
{
    int i2 = blockIdx.x * blockDim.x + threadIdx.x;  // B*H/2
    int jp = i2 & (KP - 1);
    int b  = i2 >> 8;
    int j0 = jp * 2;
    size_t e0 = (size_t)b * HH + j0;
    float h0v = sig_o[e0]     * tanhf(c_buf[e0]     * scale_c[j0]     + shift_c[j0]);
    float h1v = sig_o[e0 + 1] * tanhf(c_buf[e0 + 1] * scale_c[j0 + 1] + shift_c[j0 + 1]);
    h_buf[e0]     = h0v;
    h_buf[e0 + 1] = h1v;
    out[((size_t)b * TT + t) * HH + j0]     = h0v;
    out[((size_t)b * TT + t) * HH + j0 + 1] = h1v;
    hp_hi[i2] = (bf_hi_bits(h1v) << 16) | bf_hi_bits(h0v);
    hp_lo[i2] = (bf_lo_bits(h1v) << 16) | bf_lo_bits(h0v);
}

__global__ __launch_bounds__(256) void bnlstm_init(
    const float* __restrict__ h0, const float* __restrict__ c0,
    float* __restrict__ h_buf, float* __restrict__ c_buf,
    unsigned int* __restrict__ hp_hi, unsigned int* __restrict__ hp_lo)
{
    int i2 = blockIdx.x * blockDim.x + threadIdx.x;  // B*H/2
    if (i2 >= BB * HH / 2) return;
    int jp = i2 & (KP - 1);
    int b  = i2 >> 8;
    int j0 = jp * 2;
    float a = h0[j0], bv = h0[j0 + 1];
    size_t e0 = (size_t)b * HH + j0;
    h_buf[e0] = a;  h_buf[e0 + 1] = bv;
    c_buf[e0] = c0[j0]; c_buf[e0 + 1] = c0[j0 + 1];
    hp_hi[i2] = (bf_hi_bits(bv) << 16) | bf_hi_bits(a);
    hp_lo[i2] = (bf_lo_bits(bv) << 16) | bf_lo_bits(a);
}

__global__ __launch_bounds__(256) void bnlstm_tail(
    const float* __restrict__ h_buf, const float* __restrict__ c_buf,
    float* __restrict__ outh, float* __restrict__ outc)
{
    int i = blockIdx.x * blockDim.x + threadIdx.x;
    if (i < BB * HH) { outh[i] = h_buf[i]; outc[i] = c_buf[i]; }
}

extern "C" void kernel_launch(void* const* d_in, const int* in_sizes, int n_in,
                              void* d_out, int out_size, void* d_ws, size_t ws_size,
                              hipStream_t stream) {
    const float* input = (const float*)d_in[0];   // [B,T,F]
    const float* Wh    = (const float*)d_in[1];   // [H,4H]
    const float* Wi    = (const float*)d_in[2];   // [F,4H]
    const float* bias  = (const float*)d_in[3];
    const float* gam_h = (const float*)d_in[4];
    const float* bet_h = (const float*)d_in[5];
    const float* gam_i = (const float*)d_in[6];
    const float* bet_i = (const float*)d_in[7];
    const float* gam_c = (const float*)d_in[8];
    const float* bet_c = (const float*)d_in[9];
    const float* h0    = (const float*)d_in[10];
    const float* c0    = (const float*)d_in[11];
    float* out = (float*)d_out;                   // hiddens | h_T | c_T

    // ---- workspace layout (32-bit words)
    float* ws      = (float*)d_ws;
    float* Gh      = ws;                               // B*4H
    float* Gi      = Gh + (size_t)BB * G4;             // B*4H
    float* h_buf   = Gi + (size_t)BB * G4;             // B*H
    float* c_buf   = h_buf + (size_t)BB * HH;          // B*H
    float* sig_o   = c_buf + (size_t)BB * HH;          // B*H
    float* psum_h  = sig_o + (size_t)BB * HH;          // 2048*32
    float* psq_h   = psum_h + (size_t)G4 * 32;
    float* psum_i  = psq_h  + (size_t)G4 * 32;
    float* psq_i   = psum_i + (size_t)G4 * 32;
    float* psum_c  = psq_i  + (size_t)G4 * 32;         // 512*16
    float* psq_c   = psum_c + (size_t)HH * 16;
    float* scale_h = psq_c + (size_t)HH * 16;          // 2048 each
    float* shift_h = scale_h + G4;
    float* scale_i = shift_h + G4;
    float* shift_i = scale_i + G4;
    float* scale_c = shift_i + G4;                     // 512 each
    float* shift_c = scale_c + HH;
    unsigned int* wph_hi = (unsigned int*)(shift_c + HH);   // KP*G4 each
    unsigned int* wph_lo = wph_hi + (size_t)KP * G4;
    unsigned int* wpi_hi = wph_lo + (size_t)KP * G4;
    unsigned int* wpi_lo = wpi_hi + (size_t)KP * G4;
    unsigned int* hp_hi  = wpi_lo + (size_t)KP * G4;        // B*KP each
    unsigned int* hp_lo  = hp_hi + (size_t)BB * KP;
    size_t need = (size_t)((float*)(hp_lo + (size_t)BB * KP) - ws) * 4;
    if (ws_size < need) return;                        // deterministic guard

    const int PW = (KP * G4 + 255) / 256;
    bnlstm_packw<<<PW, 256, 0, stream>>>(Wh, wph_hi, wph_lo);
    bnlstm_packw<<<PW, 256, 0, stream>>>(Wi, wpi_hi, wpi_lo);
    bnlstm_init<<<(BB * HH / 2 + 255) / 256, 256, 0, stream>>>(
        h0, c0, h_buf, c_buf, hp_hi, hp_lo);

    const dim3 ggrid(G4 / 256, BB / 32, 2);            // (8,16,2)
    const dim3 cgrid(HH / 64, BB / 32);                // (8,16)
    const dim3 cblk(64, 4);

    for (int t = 0; t < TT; ++t) {
        bnlstm_gemm<<<ggrid, 256, 0, stream>>>(
            hp_hi, hp_lo, input + (size_t)t * FF, TT * FF,
            wph_hi, wph_lo, wpi_hi, wpi_lo,
            Gh, Gi, psum_h, psq_h, psum_i, psq_i);
        bnlstm_fin_gates<<<(2 * G4 + 255) / 256, 256, 0, stream>>>(
            psum_h, psq_h, psum_i, psq_i,
            gam_h, bet_h, gam_i, bet_i,
            scale_h, shift_h, scale_i, shift_i);
        bnlstm_cell<<<cgrid, cblk, 0, stream>>>(
            Gh, Gi, c_buf, sig_o,
            scale_h, shift_h, scale_i, shift_i, bias, psum_c, psq_c);
        bnlstm_fin_c<<<(HH + 255) / 256, 256, 0, stream>>>(
            psum_c, psq_c, gam_c, bet_c, scale_c, shift_c);
        bnlstm_hup<<<(BB * HH / 2 + 255) / 256, 256, 0, stream>>>(
            c_buf, sig_o, scale_c, shift_c, h_buf, hp_hi, hp_lo, out, t);
    }

    float* outh = out + (size_t)BB * TT * HH;
    float* outc = outh + (size_t)BB * HH;
    bnlstm_tail<<<(BB * HH + 255) / 256, 256, 0, stream>>>(h_buf, c_buf, outh, outc);
}
